// LSTMPredictor_15951508538149
// MI455X (gfx1250) — compile-verified, matrix operand order checked by reading, not run on hardware
//
#include <hip/hip_runtime.h>
#include <hip/hip_bf16.h>

#define B_TOT   256
#define T_LEN   1024
#define HID     128
#define GATES   512
#define WG_M    16
#define NBLK    (B_TOT / WG_M)   // 16 workgroups
#define NTHR    256              // 8 wave32s

typedef __attribute__((ext_vector_type(16))) __bf16 v16bf;
typedef __attribute__((ext_vector_type(8)))  float  v8f;

union Frag {
  v16bf  v;
  uint4  u[2];
  __bf16 h[16];
};

__device__ __forceinline__ float fsigm(float z) { return 1.0f / (1.0f + __expf(-z)); }
__device__ __forceinline__ float ftanh(float z) {
  z = fminf(fmaxf(z, -15.0f), 15.0f);
  float e = __expf(2.0f * z);
  return (e - 1.0f) / (e + 1.0f);
}

// Load a pre-swizzled B fragment (16 bf16 per lane, contiguous 32B) from LDS.
__device__ __forceinline__ v16bf ld_bfrag(const __bf16* p) {
  Frag f;
  f.u[0] = ((const uint4*)p)[0];
  f.u[1] = ((const uint4*)p)[1];
  return f.v;
}

// Load an A fragment (16x32 bf16 tile) from an LDS h-buffer laid out [16][128].
// A layout (ISA 7.12.2, 16-bit A 16x32): lane L: M=L%16; half=L/16 selects
// K in {8h..8h+7} (elems 0..7) and {16+8h..16+8h+7} (elems 8..15).
__device__ __forceinline__ v16bf ld_afrag(const __bf16* hbuf, int ln, int half, int k0) {
  const __bf16* p = hbuf + ln * HID + k0 + half * 8;
  Frag f;
  f.u[0] = *(const uint4*)(p);
  f.u[1] = *(const uint4*)(p + 16);
  return f.v;
}

__global__ __launch_bounds__(NTHR) void lstm2_persistent(
    const float* __restrict__ x,
    const float* __restrict__ Wih1, const float* __restrict__ Whh1,
    const float* __restrict__ bih1, const float* __restrict__ bhh1,
    const float* __restrict__ Wih2, const float* __restrict__ Whh2,
    const float* __restrict__ bih2, const float* __restrict__ bhh2,
    const float* __restrict__ Wout, const float* __restrict__ bout,
    float* __restrict__ out)
{
  // ---- LDS: layer-2 weights live here fully, pre-swizzled into B-fragment order
  __shared__ __align__(16) __bf16 s_wih2[GATES * HID];   // 128 KB
  __shared__ __align__(16) __bf16 s_whh2[GATES * HID];   // 128 KB
  __shared__ __align__(16) __bf16 s_h1[WG_M * HID];      // 4 KB
  __shared__ __align__(16) __bf16 s_h2[WG_M * HID];      // 4 KB
  __shared__ float s_x[2][WG_M];                         // x(t) double buffer
  __shared__ float s_part[8][WG_M];                      // per-wave output partials

  const int tid  = threadIdx.x;
  const int wave = tid >> 5;
  const int lane = tid & 31;
  const int half = lane >> 4;
  const int ln   = lane & 15;
  const int hc   = wave * 16 + ln;        // hidden column owned by this lane
  const int b0   = blockIdx.x * WG_M;     // batch base row

  // ---- init: swizzle W_ih2 / W_hh2 (row-major [512,128]) into bf16 B-fragments.
  // Fragment (nt,kc): B[k][n] = W[n][k], n = nt*16 + lane%16,
  // k = kc*32 + (e<8 ? 8*half+e : 16+8*half+(e-8)).  LDS offset == flat index i.
  for (int i = tid; i < GATES * HID; i += NTHR) {
    int e    = i & 15;
    int l    = (i >> 4) & 31;
    int frag = i >> 9;
    int kc   = frag & 3;
    int nt   = frag >> 2;
    int hl   = l >> 4;
    int n    = nt * 16 + (l & 15);
    int k    = kc * 32 + ((e < 8) ? (hl * 8 + e) : (16 + hl * 8 + (e - 8)));
    s_wih2[i] = (__bf16)Wih2[n * HID + k];
    s_whh2[i] = (__bf16)Whh2[n * HID + k];
  }
  for (int i = tid; i < WG_M * HID; i += NTHR) {
    s_h1[i] = (__bf16)0.0f;
    s_h2[i] = (__bf16)0.0f;
  }
  if (tid < WG_M) s_x[0][tid] = x[(size_t)(b0 + tid) * T_LEN];

  // ---- W_hh1 fragments kept register-resident (16 frags x 8 VGPRs = 128 VGPR/lane)
  Frag whh1[4][4];   // [gate][kchunk]
  {
    const int n = /* row of W_hh1 == col of W_hh1^T */ 0;
    (void)n;
  }
#pragma unroll
  for (int gi = 0; gi < 4; ++gi) {
    int n = gi * 128 + hc;
#pragma unroll
    for (int kc = 0; kc < 4; ++kc) {
#pragma unroll
      for (int e = 0; e < 16; ++e) {
        int k = kc * 32 + ((e < 8) ? (half * 8 + e) : (16 + half * 8 + (e - 8)));
        whh1[gi][kc].h[e] = (__bf16)Whh1[n * HID + k];
      }
    }
  }

  // ---- per-lane epilogue constants
  float bs1[4], wx1[4], bs2[4];
#pragma unroll
  for (int gi = 0; gi < 4; ++gi) {
    int n   = gi * 128 + hc;
    bs1[gi] = bih1[n] + bhh1[n];
    wx1[gi] = Wih1[n];           // W_ih1 is [512,1]
    bs2[gi] = bih2[n] + bhh2[n];
  }
  const float woutv = Wout[hc];
  const float boutv = bout[0];

  float c1[8], c2[8];
#pragma unroll
  for (int r = 0; r < 8; ++r) { c1[r] = 0.0f; c2[r] = 0.0f; }

  __syncthreads();

  // =========================== time loop ===========================
  for (int t = 0; t < T_LEN; ++t) {
    // prefetch x(t+1) into the other buffer (off the critical path)
    if (tid < WG_M && (t + 1) < T_LEN)
      s_x[(t + 1) & 1][tid] = x[(size_t)(b0 + tid) * T_LEN + (t + 1)];

    // ---------------- layer 1: gates1 = h1_old @ W_hh1^T ----------------
    v8f acc[4];
#pragma unroll
    for (int gi = 0; gi < 4; ++gi) acc[gi] = (v8f)0.0f;
#pragma unroll
    for (int kc = 0; kc < 4; ++kc) {
      v16bf a = ld_afrag(s_h1, ln, half, kc * 32);
#pragma unroll
      for (int gi = 0; gi < 4; ++gi)
        acc[gi] = __builtin_amdgcn_wmma_f32_16x16x32_bf16(
            false, a, false, whh1[gi][kc].v, (short)0, acc[gi], false, false);
    }

    float xv[8];
#pragma unroll
    for (int r = 0; r < 8; ++r) xv[r] = s_x[t & 1][half * 8 + r];

    float h1n[8];
#pragma unroll
    for (int r = 0; r < 8; ++r) {   // C layout: M = r + 8*half, col = hc
      float zi = acc[0][r] + bs1[0] + xv[r] * wx1[0];
      float zf = acc[1][r] + bs1[1] + xv[r] * wx1[1];
      float zg = acc[2][r] + bs1[2] + xv[r] * wx1[2];
      float zo = acc[3][r] + bs1[3] + xv[r] * wx1[3];
      float cv = fsigm(zf) * c1[r] + fsigm(zi) * ftanh(zg);
      c1[r]    = cv;
      h1n[r]   = fsigm(zo) * ftanh(cv);
    }

    __syncthreads();   // all h1_old A-fragment reads complete
#pragma unroll
    for (int r = 0; r < 8; ++r)
      s_h1[(half * 8 + r) * HID + hc] = (__bf16)h1n[r];
    __syncthreads();   // h1_new visible to all waves

    // -------- layer 2: gates2 = h1_new @ W_ih2^T + h2_old @ W_hh2^T --------
#pragma unroll
    for (int gi = 0; gi < 4; ++gi) acc[gi] = (v8f)0.0f;
#pragma unroll
    for (int kc = 0; kc < 4; ++kc) {
      v16bf a = ld_afrag(s_h1, ln, half, kc * 32);
#pragma unroll
      for (int gi = 0; gi < 4; ++gi) {
        const __bf16* wp = s_wih2 + (size_t)(((gi * 8 + wave) * 4 + kc) * 32 + lane) * 16;
        acc[gi] = __builtin_amdgcn_wmma_f32_16x16x32_bf16(
            false, a, false, ld_bfrag(wp), (short)0, acc[gi], false, false);
      }
    }
#pragma unroll
    for (int kc = 0; kc < 4; ++kc) {
      v16bf a = ld_afrag(s_h2, ln, half, kc * 32);
#pragma unroll
      for (int gi = 0; gi < 4; ++gi) {
        const __bf16* wp = s_whh2 + (size_t)(((gi * 8 + wave) * 4 + kc) * 32 + lane) * 16;
        acc[gi] = __builtin_amdgcn_wmma_f32_16x16x32_bf16(
            false, a, false, ld_bfrag(wp), (short)0, acc[gi], false, false);
      }
    }

    float h2n[8];
#pragma unroll
    for (int r = 0; r < 8; ++r) {
      float zi = acc[0][r] + bs2[0];
      float zf = acc[1][r] + bs2[1];
      float zg = acc[2][r] + bs2[2];
      float zo = acc[3][r] + bs2[3];
      float cv = fsigm(zf) * c2[r] + fsigm(zi) * ftanh(zg);
      c2[r]    = cv;
      h2n[r]   = fsigm(zo) * ftanh(cv);
    }

    __syncthreads();   // all h2_old reads complete
#pragma unroll
    for (int r = 0; r < 8; ++r) {
      s_h2[(half * 8 + r) * HID + hc] = (__bf16)h2n[r];
      // out contribution: reduce h2*W_out over the 16 columns this half-wave owns
      float v = h2n[r] * woutv;
      v += __shfl_xor(v, 1, 16);
      v += __shfl_xor(v, 2, 16);
      v += __shfl_xor(v, 4, 16);
      v += __shfl_xor(v, 8, 16);
      if (ln == 0) s_part[wave][half * 8 + r] = v;
    }
    __syncthreads();

    if (tid < WG_M) {
      float s = boutv;
#pragma unroll
      for (int w = 0; w < 8; ++w) s += s_part[w][tid];
      out[(size_t)(b0 + tid) * T_LEN + t] = s;
    }
    __syncthreads();   // protect s_part / s_x before next iteration
  }
}

extern "C" void kernel_launch(void* const* d_in, const int* in_sizes, int n_in,
                              void* d_out, int out_size, void* d_ws, size_t ws_size,
                              hipStream_t stream) {
  const float* x    = (const float*)d_in[0];
  const float* Wih1 = (const float*)d_in[1];
  const float* Whh1 = (const float*)d_in[2];
  const float* bih1 = (const float*)d_in[3];
  const float* bhh1 = (const float*)d_in[4];
  const float* Wih2 = (const float*)d_in[5];
  const float* Whh2 = (const float*)d_in[6];
  const float* bih2 = (const float*)d_in[7];
  const float* bhh2 = (const float*)d_in[8];
  const float* Wout = (const float*)d_in[9];
  const float* bout = (const float*)d_in[10];
  (void)in_sizes; (void)n_in; (void)out_size; (void)d_ws; (void)ws_size;

  lstm2_persistent<<<dim3(NBLK), dim3(NTHR), 0, stream>>>(
      x, Wih1, Whh1, bih1, bhh1, Wih2, Whh2, bih2, bhh2, Wout, bout,
      (float*)d_out);
}
